// TASFTvLLMAttentionBackend_16595753631791
// MI455X (gfx1250) — matrix operand order repaired and checked by reading, not compiled, not run on hardware
//
#include <hip/hip_runtime.h>
#include <hip/hip_bf16.h>

#define SEQ      2048
#define NH       16
#define NKVH     4
#define HD       64
#define BLK      64
#define NB       (SEQ / BLK)     // 32 blocks
#define QSTRIDE  (NH * HD)       // 1024
#define KSTRIDE  (NKVH * HD)     // 256
#define LDK      72              // padded LDS row stride (bf16 elems)
#define LDKF     68              // padded LDS row stride (f32 elems)
#define KEEP_THRESH 11468        // kept <= this  =>  sparsity >= 0.3

typedef __attribute__((ext_vector_type(16))) __bf16        v16bf;
typedef __attribute__((ext_vector_type(8)))  float          v8f;
typedef __attribute__((ext_vector_type(4)))  float          v4f;
typedef __attribute__((ext_vector_type(4)))  unsigned int   v4u;
typedef __attribute__((ext_vector_type(4)))  int            v4i;

#if defined(__has_builtin)
#  if __has_builtin(__builtin_amdgcn_global_load_async_to_lds_b128)
#    define HAS_ASYNC_LDS 1
#  endif
#endif
#ifndef HAS_ASYNC_LDS
#  define HAS_ASYNC_LDS 0
#endif

// pointer-to-int4 in global (AS1) / LDS (AS3) address spaces, per builtin sig
typedef __attribute__((address_space(1))) v4i* as1_v4i;
typedef __attribute__((address_space(3))) v4i* as3_v4i;

union FragU { v16bf b; v4u q[2]; };

// 16 consecutive bf16 from LDS (two b128 reads)
__device__ __forceinline__ v16bf lds_read16(const __bf16* p) {
  FragU r;
  r.q[0] = *(const v4u*)(p);
  r.q[1] = *(const v4u*)(p + 8);
  return r.b;
}
// two disjoint runs of 8 bf16 (A-fragment pattern)
__device__ __forceinline__ v16bf lds_read8x2(const __bf16* p0, const __bf16* p1) {
  FragU r;
  r.q[0] = *(const v4u*)(p0);
  r.q[1] = *(const v4u*)(p1);
  return r.b;
}
// 16 consecutive f32 -> bf16 fragment (native hw converts)
__device__ __forceinline__ v16bf cvt16(const float* p) {
  v4f x0 = *(const v4f*)(p);
  v4f x1 = *(const v4f*)(p + 4);
  v4f x2 = *(const v4f*)(p + 8);
  v4f x3 = *(const v4f*)(p + 12);
  FragU r;
#pragma unroll
  for (int e = 0; e < 4; ++e) {
    r.b[e]      = (__bf16)x0[e];
    r.b[4 + e]  = (__bf16)x1[e];
    r.b[8 + e]  = (__bf16)x2[e];
    r.b[12 + e] = (__bf16)x3[e];
  }
  return r.b;
}

__device__ __forceinline__ v8f wmma_bf16(v16bf a, v16bf b, v8f c) {
  return __builtin_amdgcn_wmma_f32_16x16x32_bf16(false, a, false, b,
                                                 (short)0, c, false, false);
}

// ---- DPP16 ROW_XMASK cross-lane reduction within each 16-lane row ----
template <int CTRL>
__device__ __forceinline__ float dpp_mov(float x) {
  return __int_as_float(
      __builtin_amdgcn_update_dpp(0, __float_as_int(x), CTRL, 0xf, 0xf, true));
}
__device__ __forceinline__ float row_reduce_max(float x) {
  x = fmaxf(x, dpp_mov<0x161>(x));   // xor 1
  x = fmaxf(x, dpp_mov<0x162>(x));   // xor 2
  x = fmaxf(x, dpp_mov<0x164>(x));   // xor 4
  x = fmaxf(x, dpp_mov<0x168>(x));   // xor 8
  return x;
}
__device__ __forceinline__ float row_reduce_add(float x) {
  x += dpp_mov<0x161>(x);
  x += dpp_mov<0x162>(x);
  x += dpp_mov<0x164>(x);
  x += dpp_mov<0x168>(x);
  return x;
}

// ---------------------------------------------------------------------------
// Gate: per head, block mean-pool + bilinear score -> block mask + kept count
// sigmoid(x) > 0.5  <=>  x > 0, so no transcendental needed.
// ---------------------------------------------------------------------------
__global__ __launch_bounds__(128) void gate_kernel(
    const float* __restrict__ Q, const float* __restrict__ K,
    const float* __restrict__ W, unsigned char* __restrict__ gmask,
    int* __restrict__ kept)
{
  __shared__ float qm[NB * HD];
  __shared__ float km[NB * HD];
  __shared__ float tm[NB * HD];
  const int h   = blockIdx.x;
  const int kh  = h >> 2;                       // GQA: repeat_interleave(4)
  const int tid = threadIdx.x;

  for (int idx = tid; idx < NB * HD; idx += 128) {
    const int b = idx >> 6, d = idx & 63;
    float sq = 0.f, sk = 0.f;
    for (int t = 0; t < BLK; ++t) {
      sq += Q[(size_t)(b * BLK + t) * QSTRIDE + h  * HD + d];
      sk += K[(size_t)(b * BLK + t) * KSTRIDE + kh * HD + d];
    }
    qm[idx] = sq * (1.f / BLK);
    km[idx] = sk * (1.f / BLK);
  }
  __syncthreads();

  // tm = qm @ W[h]  (32x64 @ 64x64)
  for (int idx = tid; idx < NB * HD; idx += 128) {
    const int b = idx >> 6, e = idx & 63;
    float acc = 0.f;
    const float* wh = W + (size_t)h * HD * HD + e;
#pragma unroll 8
    for (int d = 0; d < HD; ++d) acc += qm[(b << 6) + d] * wh[d * HD];
    tm[idx] = acc;
  }
  __syncthreads();

  int cnt = 0;
  for (int idx = tid; idx < NB * NB; idx += 128) {
    const int qi = idx >> 5, ki = idx & 31;
    unsigned char kp = 0;
    if (ki <= qi) {                              // block-causal tril
      if (ki == qi) {
        kp = 1;                                  // diagonal always kept
      } else {
        float acc = 0.f;
#pragma unroll 8
        for (int e = 0; e < HD; ++e) acc += tm[(qi << 6) + e] * km[(ki << 6) + e];
        kp = (acc > 0.f) ? 1 : 0;                // sigmoid(acc*scale) > 0.5
      }
      cnt += kp;
    }
    gmask[h * NB * NB + idx] = kp;
  }
#pragma unroll
  for (int off = 1; off < 32; off <<= 1) cnt += __shfl_xor(cnt, off, 32);
  if ((tid & 31) == 0) atomicAdd(kept, cnt);
}

__global__ void zero_kernel(int* p) {
  if (threadIdx.x == 0 && blockIdx.x == 0) *p = 0;
}

// ---------------------------------------------------------------------------
// Flash attention with WMMA bf16. 4 waves/WG; wave w owns query rows
// [qb*64 + w*16, +16). K tile staged raw-f32 via async copy-to-LDS
// (ASYNCcnt), V staged transposed-bf16 manually, P re-laid out via LDS.
// ---------------------------------------------------------------------------
__global__ __launch_bounds__(128) void attn_kernel(
    const float* __restrict__ Q, const float* __restrict__ K,
    const float* __restrict__ V, const unsigned char* __restrict__ gmask,
    const int* __restrict__ keptp, float* __restrict__ out)
{
  __shared__ float  sKf[BLK * LDKF];             // K tile raw f32: [key][d]
  __shared__ __bf16 sV[HD * LDK];                // V tile transposed: [d][key]
  __shared__ __bf16 sP[4 * 16 * LDK];            // per-wave P re-layout strip

  const int qb   = blockIdx.x;
  const int h    = blockIdx.y;
  const int kh   = h >> 2;
  const int tid  = threadIdx.x;
  const int wave = tid >> 5;
  const int lane = tid & 31;
  const int ln   = lane & 15;                    // N / M index within half
  const int hi   = lane >> 4;                    // lane-half selector

  const bool dense = (*keptp) > KEEP_THRESH;     // sparsity < 0.3 -> dense, no mask
  const int  jend  = dense ? (NB - 1) : qb;

  // ---- Q A-fragments (16x32 bf16 layout: lane M = ln, K split by hi) ----
  const float* qrow = Q + (size_t)(qb * BLK + wave * 16 + ln) * QSTRIDE + h * HD;
  v16bf a0, a1;
  {
    FragU ua;
#pragma unroll
    for (int e = 0; e < 8; ++e) ua.b[e]     = (__bf16)qrow[hi * 16 + e];
#pragma unroll
    for (int e = 0; e < 8; ++e) ua.b[8 + e] = (__bf16)qrow[8 + hi * 16 + e];
    a0 = ua.b;
#pragma unroll
    for (int e = 0; e < 8; ++e) ua.b[e]     = (__bf16)qrow[32 + hi * 16 + e];
#pragma unroll
    for (int e = 0; e < 8; ++e) ua.b[8 + e] = (__bf16)qrow[40 + hi * 16 + e];
    a1 = ua.b;
  }

  float m_r[8], l_r[8];
#pragma unroll
  for (int vv = 0; vv < 8; ++vv) { m_r[vv] = -1e30f; l_r[vv] = 0.f; }
  v8f o[4];
#pragma unroll
  for (int t = 0; t < 4; ++t) o[t] = v8f{};

  __bf16* pw = &sP[wave * 16 * LDK];

  for (int j = 0; j <= jend; ++j) {
    // workgroup-uniform skip decision (depends only on h, qb, j)
    const bool keep = dense || (j == qb) || (gmask[(h * NB + qb) * NB + j] != 0);
    if (!keep) continue;

    if (j < jend)  // pull next V tile toward L2/WGP$ (global_prefetch_b8)
      __builtin_prefetch(V + (size_t)((j + 1) * BLK + (tid >> 1)) * KSTRIDE +
                             kh * HD + (tid & 1) * 32, 0, 1);

    __syncthreads();

    // ---- K tile: async copy raw f32 global -> LDS (ASYNCcnt tracked) ----
#pragma unroll 2
    for (int it = tid; it < BLK * 16; it += 128) {
      const int row = it >> 4;                   // key within tile
      const int c4  = (it & 15) << 2;            // dim base
      const float* gp = K + (size_t)(j * BLK + row) * KSTRIDE + kh * HD + c4;
      float* lp = &sKf[row * LDKF + c4];
#if HAS_ASYNC_LDS
      __builtin_amdgcn_global_load_async_to_lds_b128((as1_v4i)gp, (as3_v4i)lp, 0, 0);
#else
      *(v4f*)lp = *(const v4f*)gp;
#endif
    }

    // ---- V tile: transpose + convert to bf16 (needs element shuffle) ----
#pragma unroll 2
    for (int it = tid; it < BLK * 16; it += 128) {
      const int row = it >> 4;
      const int c4  = (it & 15) << 2;
      const v4f v4 = *(const v4f*)(V + (size_t)(j * BLK + row) * KSTRIDE +
                                   kh * HD + c4);
      sV[(c4 + 0) * LDK + row] = (__bf16)v4.x;
      sV[(c4 + 1) * LDK + row] = (__bf16)v4.y;
      sV[(c4 + 2) * LDK + row] = (__bf16)v4.z;
      sV[(c4 + 3) * LDK + row] = (__bf16)v4.w;
    }

#if HAS_ASYNC_LDS
#  if __has_builtin(__builtin_amdgcn_s_wait_asynccnt)
    __builtin_amdgcn_s_wait_asynccnt(0);
#  else
    asm volatile("s_wait_asynccnt 0" ::: "memory");
#  endif
#endif
    __syncthreads();

    // ---- S = Q @ K^T : 4 tiles of 16x16, 2 WMMA each (K-dim 64) ----
    v8f s[4];
#pragma unroll
    for (int n = 0; n < 4; ++n) {
      const float* kp = &sKf[(16 * n + ln) * LDKF + hi * 16];
      v16bf b0 = cvt16(kp);
      v16bf b1 = cvt16(kp + 32);
      v8f acc = v8f{};
      acc = wmma_bf16(a0, b0, acc);
      acc = wmma_bf16(a1, b1, acc);
      s[n] = acc;
    }

    // ---- online softmax (per-row state, DPP16 row_xmask butterflies) ----
    const bool diag = (!dense) && (j == qb);     // token-causal on diagonal
    float alpha[8];
#pragma unroll
    for (int vv = 0; vv < 8; ++vv) {
      const int qloc = wave * 16 + vv + hi * 8;  // query pos within 64-block
      float mx = m_r[vv];
#pragma unroll
      for (int n = 0; n < 4; ++n) {
        float x = s[n][vv] * 0.125f;             // 1/sqrt(64)
        if (diag && (16 * n + ln > qloc)) x = -1e30f;
        s[n][vv] = x;
        mx = fmaxf(mx, x);
      }
      mx = row_reduce_max(mx);
      float sum = 0.f;
#pragma unroll
      for (int n = 0; n < 4; ++n) {
        const float p = __expf(s[n][vv] - mx);
        s[n][vv] = p;
        sum += p;
      }
      sum = row_reduce_add(sum);
      const float a = __expf(m_r[vv] - mx);
      alpha[vv] = a;
      l_r[vv]   = l_r[vv] * a + sum;
      m_r[vv]   = mx;
    }

    // ---- re-layout P: C-layout regs -> per-wave LDS strip -> A fragments ----
#pragma unroll
    for (int n = 0; n < 4; ++n)
#pragma unroll
      for (int vv = 0; vv < 8; ++vv)
        pw[(vv + hi * 8) * LDK + 16 * n + ln] = (__bf16)s[n][vv];

    const __bf16* pr = pw + ln * LDK;
    v16bf p0 = lds_read8x2(pr + hi * 16,      pr + 8 + hi * 16);
    v16bf p1 = lds_read8x2(pr + 32 + hi * 16, pr + 40 + hi * 16);

    // ---- O = O*alpha + P @ V : 4 tiles, 2 WMMA each ----
#pragma unroll
    for (int t = 0; t < 4; ++t) {
#pragma unroll
      for (int vv = 0; vv < 8; ++vv) o[t][vv] *= alpha[vv];
      const __bf16* vp = &sV[(16 * t + ln) * LDK + hi * 16];
      v16bf bv0 = lds_read16(vp);
      v16bf bv1 = lds_read16(vp + 32);
      o[t] = wmma_bf16(p0, bv0, o[t]);
      o[t] = wmma_bf16(p1, bv1, o[t]);
    }
  }

  // ---- normalize and store [S, H*D] ----
#pragma unroll
  for (int t = 0; t < 4; ++t) {
#pragma unroll
    for (int vv = 0; vv < 8; ++vv) {
      const int row = qb * BLK + wave * 16 + vv + hi * 8;
      out[(size_t)row * QSTRIDE + h * HD + 16 * t + ln] = o[t][vv] / l_r[vv];
    }
  }
}

extern "C" void kernel_launch(void* const* d_in, const int* in_sizes, int n_in,
                              void* d_out, int out_size, void* d_ws, size_t ws_size,
                              hipStream_t stream) {
  (void)in_sizes; (void)n_in; (void)out_size; (void)ws_size;
  const float* Q = (const float*)d_in[0];
  const float* K = (const float*)d_in[1];
  const float* V = (const float*)d_in[2];
  const float* W = (const float*)d_in[3];
  float* out = (float*)d_out;

  int* kept            = (int*)d_ws;                     // [0,4): kept-block count
  unsigned char* gmask = (unsigned char*)d_ws + 64;      // 16*32*32 mask bytes

  zero_kernel<<<1, 32, 0, stream>>>(kept);
  gate_kernel<<<NH, 128, 0, stream>>>(Q, K, W, gmask, kept);
  attn_kernel<<<dim3(NB, NH), 128, 0, stream>>>(Q, K, V, gmask, kept, out);
}
